// ImageLinearAttention_7997229105327
// MI455X (gfx1250) — compile-verified, matrix operand order checked by reading, not run on hardware
//
#include <hip/hip_runtime.h>
#include <hip/hip_bf16.h>
#include <math.h>

// MI455X / gfx1250, wave32. bf16 WMMA + async global->LDS (ASYNCcnt) paths.
// Problem: B=32, C=256, N=4096, HEADS=8, KD=VD=64.

typedef __attribute__((ext_vector_type(16))) __bf16 bf16x16;
typedef __attribute__((ext_vector_type(8)))  float  floatx8;

#define LDT 48                      // LDS row stride (bf16): 96B, 16B-aligned
#define A_BYTES (128 * LDT * 2)     // 12288 B per staged tile
#define B_OFF   A_BYTES
#define T_STRIDE 132
#define INV_OFF (128 * T_STRIDE * 4)        // 67584
#define PROJ_SMEM (INV_OFF + 2 * 128 * 4)   // 68608 (epilogue view dominates)
#define OUTP_BUF  (2 * A_BYTES)             // 24576 (A+B per buffer)
#define OUTP_SMEM (2 * OUTP_BUF)            // 49152 (double-buffered)

// ---- CDNA5 async global->LDS copy of 16 bytes per lane (ASYNCcnt) ----------
__device__ __forceinline__ void async_b128(unsigned lds_off, const void* gp) {
  asm volatile("global_load_async_to_lds_b128 %0, %1, off"
               :: "v"(lds_off), "v"(gp) : "memory");
}
__device__ __forceinline__ void wait_async0() {
#if __has_builtin(__builtin_amdgcn_s_wait_asynccnt)
  __builtin_amdgcn_s_wait_asynccnt(0);
#else
  asm volatile("s_wait_asynccnt 0" ::: "memory");
#endif
}

// Pack two f32 -> one dword of 2 bf16 (lo = first). Lowers to v_cvt_pk_bf16_f32.
__device__ __forceinline__ unsigned pk_bf16(float lo, float hi) {
  union { __bf16 h[2]; unsigned u; } t;
  t.h[0] = (__bf16)lo; t.h[1] = (__bf16)hi;
  return t.u;
}

// Stage a 128x32 bf16 tile (rows contiguous in memory, rowstride elems apart)
// into LDS at ldsoff with row stride LDT, via async b128 copies (2 per thread).
__device__ __forceinline__ void stage_tile_async(const __bf16* src, int rowstride,
                                                 unsigned ldsoff, int tid) {
#pragma unroll
  for (int i = 0; i < 2; ++i) {
    int idx = i * 256 + tid;
    int r = idx >> 2, q4 = idx & 3;
    async_b128(ldsoff + (unsigned)((r * LDT + q4 * 8) * 2),
               src + (size_t)r * rowstride + q4 * 8);
  }
}

// Load one 16x32 bf16 A/B fragment from LDS (rows of 32 K, stride LDT elems).
// Layout per ISA 7.12.2: dword j<4 -> K pairs (2j)+8*half, j>=4 -> 16+...
__device__ __forceinline__ bf16x16 ld_frag(const __bf16* base, int lane) {
  const unsigned* p = reinterpret_cast<const unsigned*>(base);
  const int hf = (lane >> 4) & 1;
  union { bf16x16 v; unsigned u[8]; } r;
#pragma unroll
  for (int j = 0; j < 4; ++j) {
    r.u[j]     = p[4 * hf + j];
    r.u[4 + j] = p[8 + 4 * hf + j];
  }
  return r.v;
}

// ---------------------------------------------------------------------------
// Kernel 0: pack [Wq;Wk;Wv] -> contiguous bf16 640x256 + packed bias (one-shot)
// ---------------------------------------------------------------------------
__global__ __launch_bounds__(256)
void pack_w_kernel(const float* __restrict__ Wq, const float* __restrict__ bq,
                   const float* __restrict__ Wk, const float* __restrict__ bk,
                   const float* __restrict__ Wv, const float* __restrict__ bv,
                   __bf16* __restrict__ wqkv, float* __restrict__ biasp)
{
  int i = blockIdx.x * 256 + threadIdx.x;       // 640*256 = 163840
  if (i < 640 * 256) {
    int row = i >> 8, col = i & 255;
    float w;
    if (row < 512)      w = Wq[row * 256 + col];
    else if (row < 576) w = Wk[(row - 512) * 256 + col];
    else                w = Wv[(row - 576) * 256 + col];
    wqkv[i] = (__bf16)w;
  }
  if (i < 640)
    biasp[i] = (i < 512) ? bq[i] : (i < 576 ? bk[i - 512] : bv[i - 576]);
}

// ---------------------------------------------------------------------------
// Kernel 1: fused QKV projection, 640x256 @ 256x128 per (batch, col-block).
// A-tile async global->LDS; B-tile: paired f32 loads -> v_cvt_pk_bf16_f32 ->
// ds_store_b32. q written TRANSPOSED (qt[b][n][512], K contiguous) so the
// final GEMM can stage its B-tile asynchronously too.
// ---------------------------------------------------------------------------
__global__ __launch_bounds__(256)
void proj_qkv_kernel(const float* __restrict__ x,
                     const __bf16* __restrict__ wqkv, const float* __restrict__ biasp,
                     __bf16* __restrict__ qt, float* __restrict__ kbuf,
                     __bf16* __restrict__ vbuf)
{
  extern __shared__ char smem[];
  __bf16* Bs = (__bf16*)(smem + B_OFF);    // [128 cols][LDT] ([col][k])

  const int rb  = blockIdx.x;              // 0..4
  const int n0  = blockIdx.y * 128;
  const int b   = blockIdx.z;
  const int tid = threadIdx.x, lane = tid & 31, wave = tid >> 5;
  const int wr = wave & 3, wc = wave >> 2;

  floatx8 zero = {};
  floatx8 acc[2][4];
#pragma unroll
  for (int i = 0; i < 2; ++i)
#pragma unroll
    for (int j = 0; j < 4; ++j) acc[i][j] = zero;

  const size_t xb = (size_t)b * 256 * 4096;

  for (int kk = 0; kk < 256; kk += 32) {
    // ---- A: 128x32 bf16 weight tile via async copies
    stage_tile_async(wqkv + (size_t)rb * 128 * 256 + kk, 256, 0, tid);
    // ---- B: all global loads first (one loadcnt wait), then pack+store
    float4 r0[2], r1[2];
#pragma unroll
    for (int i = 0; i < 2; ++i) {
      int idx = i * 256 + tid;
      int k2 = idx >> 5, c4 = (idx & 31) * 4;
      const float* xp = x + xb + (size_t)(kk + 2 * k2) * 4096 + n0 + c4;
      r0[i] = *reinterpret_cast<const float4*>(xp);
      r1[i] = *reinterpret_cast<const float4*>(xp + 4096);
    }
#pragma unroll
    for (int i = 0; i < 2; ++i) {
      int idx = i * 256 + tid;
      int k2 = idx >> 5, c4 = (idx & 31) * 4;
      *reinterpret_cast<unsigned*>(&Bs[(c4 + 0) * LDT + 2 * k2]) = pk_bf16(r0[i].x, r1[i].x);
      *reinterpret_cast<unsigned*>(&Bs[(c4 + 1) * LDT + 2 * k2]) = pk_bf16(r0[i].y, r1[i].y);
      *reinterpret_cast<unsigned*>(&Bs[(c4 + 2) * LDT + 2 * k2]) = pk_bf16(r0[i].z, r1[i].z);
      *reinterpret_cast<unsigned*>(&Bs[(c4 + 3) * LDT + 2 * k2]) = pk_bf16(r0[i].w, r1[i].w);
    }
    if (kk + 32 < 256)   // global_prefetch_b8 for the next x strip
      __builtin_prefetch(x + xb + (size_t)(kk + 32) * 4096 + n0 + (tid & 127), 0, 1);
    wait_async0();
    __syncthreads();

    const __bf16* As = (const __bf16*)smem;
    bf16x16 af[2], bfr[4];
#pragma unroll
    for (int rt = 0; rt < 2; ++rt)
      af[rt] = ld_frag(As + (wr * 32 + rt * 16 + (lane & 15)) * LDT, lane);
#pragma unroll
    for (int ct = 0; ct < 4; ++ct)
      bfr[ct] = ld_frag(Bs + (wc * 64 + ct * 16 + (lane & 15)) * LDT, lane);
#pragma unroll
    for (int rt = 0; rt < 2; ++rt)
#pragma unroll
      for (int ct = 0; ct < 4; ++ct)
        acc[rt][ct] = __builtin_amdgcn_wmma_f32_16x16x32_bf16(
            false, af[rt], false, bfr[ct], (short)0, acc[rt][ct], false, false);
    __syncthreads();
  }

  // ---- epilogue: bias add, stash f32 tile in LDS (reuse staging space) ----
  float* t   = (float*)smem;                  // [128][T_STRIDE]
  float* inv = (float*)(smem + INV_OFF);      // [2][128]
  const int hf = lane >> 4, cn = lane & 15;
#pragma unroll
  for (int rt = 0; rt < 2; ++rt)
#pragma unroll
    for (int ct = 0; ct < 4; ++ct)
#pragma unroll
      for (int r = 0; r < 8; ++r) {
        int row = wr * 32 + rt * 16 + hf * 8 + r;
        int col = wc * 64 + ct * 16 + cn;
        t[row * T_STRIDE + col] = acc[rt][ct][r] + biasp[rb * 128 + row];
      }
  __syncthreads();

  if (rb < 4) {
    // local softmax over d=64 (one head per 64-row segment), per column
    int seg = tid >> 7, col = tid & 127;
    float m = -1e30f;
    for (int d = 0; d < 64; ++d) m = fmaxf(m, t[(seg * 64 + d) * T_STRIDE + col]);
    float s = 0.f;
    for (int d = 0; d < 64; ++d) {
      float e2 = __expf(t[(seg * 64 + d) * T_STRIDE + col] - m);
      t[(seg * 64 + d) * T_STRIDE + col] = e2;
      s += e2;
    }
    inv[seg * 128 + col] = 1.0f / s;
    __syncthreads();
    // write q TRANSPOSED: qt[(b*4096 + n0+col)*512 + rb*128 + row], 16B stores
    size_t qtb = ((size_t)b * 4096 + n0) * 512 + (size_t)rb * 128;
    for (int i = 0; i < 8; ++i) {
      int idx = i * 256 + tid;
      int col = idx & 127, r0 = (idx >> 7) * 8;       // 16 row-chunks of 8
      float iv = inv[(r0 >> 6) * 128 + col];
      union { uint4 u; unsigned w[4]; } o;
#pragma unroll
      for (int j = 0; j < 4; ++j)
        o.w[j] = pk_bf16(t[(r0 + 2 * j) * T_STRIDE + col] * iv,
                         t[(r0 + 2 * j + 1) * T_STRIDE + col] * iv);
      *reinterpret_cast<uint4*>(qt + qtb + (size_t)col * 512 + r0) = o.u;
    }
  } else {
    // rows 0..63 = k (raw f32 for global softmax), rows 64..127 = v (bf16)
    size_t kb = (size_t)b * 64 * 4096 + n0;
    for (int i = 0; i < 64; ++i) {
      int idx = i * 256 + tid;
      int row = idx >> 7, c2 = idx & 127;
      float v = t[row * T_STRIDE + c2];
      if (row < 64) kbuf[kb + (size_t)row * 4096 + c2] = v;
      else          vbuf[kb + (size_t)(row - 64) * 4096 + c2] = (__bf16)v;
    }
  }
}

// ---------------------------------------------------------------------------
// Kernel 2: per (b,d) row: max over n and sum of exp (k softmax statistics).
// ---------------------------------------------------------------------------
__global__ __launch_bounds__(256)
void kstats_kernel(const float* __restrict__ kbuf,
                   float* __restrict__ kmax, float* __restrict__ ksum)
{
  __shared__ float red[256];
  const int row = blockIdx.x;
  const float4* p4 = (const float4*)(kbuf + (size_t)row * 4096);
  const int tid = threadIdx.x;
  float m = -1e30f;
  for (int i = tid; i < 1024; i += 256) {
    float4 v = p4[i];
    m = fmaxf(m, fmaxf(fmaxf(v.x, v.y), fmaxf(v.z, v.w)));
  }
  red[tid] = m; __syncthreads();
  for (int s = 128; s > 0; s >>= 1) {
    if (tid < s) red[tid] = fmaxf(red[tid], red[tid + s]);
    __syncthreads();
  }
  m = red[0]; __syncthreads();
  float s = 0.f;
  for (int i = tid; i < 1024; i += 256) {
    float4 v = p4[i];
    s += __expf(v.x - m) + __expf(v.y - m) + __expf(v.z - m) + __expf(v.w - m);
  }
  red[tid] = s; __syncthreads();
  for (int st = 128; st > 0; st >>= 1) {
    if (tid < st) red[tid] += red[tid + st];
    __syncthreads();
  }
  if (tid == 0) { kmax[row] = m; ksum[row] = red[0]; }
}

// ---------------------------------------------------------------------------
// Kernel 3: ctx[b] (64x64) = exp(k - max) @ v^T / sum, K = 4096.
// grid 32 (one per batch), 128 threads (4 waves; wave = d-tile).
// ---------------------------------------------------------------------------
__global__ __launch_bounds__(128)
void ctx_kernel(const float* __restrict__ kbuf, const __bf16* __restrict__ vbuf,
                const float* __restrict__ kmax, const float* __restrict__ ksum,
                float* __restrict__ ctxb)
{
  __shared__ __bf16 A[64][LDT];
  __shared__ __bf16 Bm[64][LDT];
  const int b = blockIdx.x;
  const int tid = threadIdx.x, lane = tid & 31, wave = tid >> 5;
  floatx8 zero = {};
  floatx8 acc[4];
#pragma unroll
  for (int c = 0; c < 4; ++c) acc[c] = zero;
  const size_t base = (size_t)b * 64 * 4096;

  for (int nn = 0; nn < 4096; nn += 32) {
#pragma unroll
    for (int i = 0; i < 8; ++i) {
      int idx = i * 128 + tid;            // 1024 dword pairs
      int d = idx >> 4, k2 = (idx & 15) * 2;
      float km = kmax[b * 64 + d];
      const float* kp = kbuf + base + (size_t)d * 4096 + nn + k2;
      *reinterpret_cast<unsigned*>(&A[d][k2]) =
          pk_bf16(__expf(kp[0] - km), __expf(kp[1] - km));
      *reinterpret_cast<unsigned*>(&Bm[d][k2]) =
          *reinterpret_cast<const unsigned*>(vbuf + base + (size_t)d * 4096 + nn + k2);
    }
    __syncthreads();
    bf16x16 af = ld_frag(&A[wave * 16 + (lane & 15)][0], lane);
#pragma unroll
    for (int ct = 0; ct < 4; ++ct) {
      bf16x16 bfr = ld_frag(&Bm[ct * 16 + (lane & 15)][0], lane);
      acc[ct] = __builtin_amdgcn_wmma_f32_16x16x32_bf16(
          false, af, false, bfr, (short)0, acc[ct], false, false);
    }
    __syncthreads();
  }

  const int hf = lane >> 4, cn = lane & 15;
#pragma unroll
  for (int ct = 0; ct < 4; ++ct)
#pragma unroll
    for (int r = 0; r < 8; ++r) {
      int d = wave * 16 + hf * 8 + r;
      int e = ct * 16 + cn;
      ctxb[(size_t)b * 64 * 64 + d * 64 + e] = acc[ct][r] / ksum[b * 64 + d];
    }
}

// ---------------------------------------------------------------------------
// Kernel 4: Wc[b][:, h*64+d] = Wo[:, h*64+e] @ ctx[b]^T  (256x64 @ 64x64).
// grid (8 heads, 32 batches), 256 threads (8 waves).
// ---------------------------------------------------------------------------
__global__ __launch_bounds__(256)
void wc_kernel(const float* __restrict__ Wo, const float* __restrict__ ctxb,
               __bf16* __restrict__ wcb)
{
  __shared__ __bf16 As[256][72];
  __shared__ __bf16 Bs[64][72];
  const int h = blockIdx.x, b = blockIdx.y;
  const int tid = threadIdx.x, lane = tid & 31, wave = tid >> 5;

  for (int i = 0; i < 32; ++i) {          // 8192 Wo pairs
    int idx = i * 256 + tid;
    int o = idx >> 5, e2 = (idx & 31) * 2;
    const float* wp = Wo + (size_t)o * 512 + h * 64 + e2;
    *reinterpret_cast<unsigned*>(&As[o][e2]) = pk_bf16(wp[0], wp[1]);
  }
  for (int i = 0; i < 8; ++i) {           // 2048 ctx pairs
    int idx = i * 256 + tid;
    int d = idx >> 5, e2 = (idx & 31) * 2;
    const float* cp = ctxb + (size_t)b * 4096 + d * 64 + e2;
    *reinterpret_cast<unsigned*>(&Bs[d][e2]) = pk_bf16(cp[0], cp[1]);
  }
  __syncthreads();

  floatx8 zero = {};
  floatx8 acc[2][4];
#pragma unroll
  for (int i = 0; i < 2; ++i)
#pragma unroll
    for (int j = 0; j < 4; ++j) acc[i][j] = zero;

#pragma unroll
  for (int ks = 0; ks < 64; ks += 32) {
    bf16x16 af[2], bfr[4];
#pragma unroll
    for (int rt = 0; rt < 2; ++rt)
      af[rt] = ld_frag(&As[wave * 32 + rt * 16 + (lane & 15)][ks], lane);
#pragma unroll
    for (int ct = 0; ct < 4; ++ct)
      bfr[ct] = ld_frag(&Bs[ct * 16 + (lane & 15)][ks], lane);
#pragma unroll
    for (int rt = 0; rt < 2; ++rt)
#pragma unroll
      for (int ct = 0; ct < 4; ++ct)
        acc[rt][ct] = __builtin_amdgcn_wmma_f32_16x16x32_bf16(
            false, af[rt], false, bfr[ct], (short)0, acc[rt][ct], false, false);
  }

  const int hf = lane >> 4, cn = lane & 15;
#pragma unroll
  for (int rt = 0; rt < 2; ++rt)
#pragma unroll
    for (int ct = 0; ct < 4; ++ct)
#pragma unroll
      for (int r = 0; r < 8; ++r) {
        int o = wave * 32 + rt * 16 + hf * 8 + r;
        int d = ct * 16 + cn;
        wcb[(size_t)b * 256 * 512 + (size_t)o * 512 + h * 64 + d] =
            (__bf16)acc[rt][ct][r];
      }
}

// ---------------------------------------------------------------------------
// Kernel 5: out[b] = Wc[b] (256x512) @ qt[b]^T + bo.  grid (2,32,32).
// BOTH tiles staged via async global->LDS, double-buffered: copies for tile
// i+1 are in flight while tile i's WMMAs execute (ASYNCcnt overlap).
// ---------------------------------------------------------------------------
__global__ __launch_bounds__(256)
void out_proj_kernel(const __bf16* __restrict__ wcb, const __bf16* __restrict__ qt,
                     const float* __restrict__ bo, float* __restrict__ out)
{
  extern __shared__ char smem[];
  const int rb = blockIdx.x;           // 0..1
  const int n0 = blockIdx.y * 128;
  const int b  = blockIdx.z;
  const int tid = threadIdx.x, lane = tid & 31, wave = tid >> 5;
  const int wr = wave & 3, wc = wave >> 2;

  const __bf16* abase = wcb + (size_t)b * 131072 + (size_t)rb * 128 * 512;
  const __bf16* bbase = qt + ((size_t)b * 4096 + n0) * 512;

  floatx8 zero = {};
  floatx8 acc[2][4];
#pragma unroll
  for (int i = 0; i < 2; ++i)
#pragma unroll
    for (int j = 0; j < 4; ++j) acc[i][j] = zero;

  // prologue: stage tile 0 into buffer 0
  stage_tile_async(abase, 512, 0, tid);
  stage_tile_async(bbase, 512, A_BYTES, tid);
  wait_async0();
  __syncthreads();

  for (int kk = 0; kk < 512; kk += 32) {
    const unsigned co = ((kk >> 5) & 1) * (unsigned)OUTP_BUF;
    const unsigned po = (unsigned)OUTP_BUF - co;   // other buffer
    if (kk + 32 < 512) {                           // prefetch next tile (async)
      stage_tile_async(abase + kk + 32, 512, po, tid);
      stage_tile_async(bbase + kk + 32, 512, po + A_BYTES, tid);
    }
    const __bf16* As = (const __bf16*)(smem + co);
    const __bf16* Bs = (const __bf16*)(smem + co + A_BYTES);
    bf16x16 af[2], bfr[4];
#pragma unroll
    for (int rt = 0; rt < 2; ++rt)
      af[rt] = ld_frag(As + (wr * 32 + rt * 16 + (lane & 15)) * LDT, lane);
#pragma unroll
    for (int ct = 0; ct < 4; ++ct)
      bfr[ct] = ld_frag(Bs + (wc * 64 + ct * 16 + (lane & 15)) * LDT, lane);
#pragma unroll
    for (int rt = 0; rt < 2; ++rt)
#pragma unroll
      for (int ct = 0; ct < 4; ++ct)
        acc[rt][ct] = __builtin_amdgcn_wmma_f32_16x16x32_bf16(
            false, af[rt], false, bfr[ct], (short)0, acc[rt][ct], false, false);
    if (kk + 32 < 512) {
      wait_async0();       // next buffer fully written (own wave's copies)
      __syncthreads();     // ... and everyone's; also fences this tile's reads
    }
  }

  const int hf = lane >> 4, cn = lane & 15;
  const size_t ob = (size_t)b * 256 * 4096 + n0;
#pragma unroll
  for (int rt = 0; rt < 2; ++rt)
#pragma unroll
    for (int ct = 0; ct < 4; ++ct)
#pragma unroll
      for (int r = 0; r < 8; ++r) {
        int row = wr * 32 + rt * 16 + hf * 8 + r;
        int col = wc * 64 + ct * 16 + cn;
        int grow = rb * 128 + row;
        out[ob + (size_t)grow * 4096 + col] = acc[rt][ct][r] + bo[grow];
      }
}

// ---------------------------------------------------------------------------
extern "C" void kernel_launch(void* const* d_in, const int* in_sizes, int n_in,
                              void* d_out, int out_size, void* d_ws, size_t ws_size,
                              hipStream_t stream) {
  const float* x  = (const float*)d_in[0];
  const float* Wq = (const float*)d_in[1];
  const float* bq = (const float*)d_in[2];
  const float* Wk = (const float*)d_in[3];
  const float* bk = (const float*)d_in[4];
  const float* Wv = (const float*)d_in[5];
  const float* bv = (const float*)d_in[6];
  const float* Wo = (const float*)d_in[7];
  const float* bo = (const float*)d_in[8];
  float* out = (float*)d_out;

  // Workspace carve-up (~195 MB total).
  char* p = (char*)d_ws;
  auto take = [&](size_t bytes) {
    char* r = p; p += (bytes + 255) & ~(size_t)255; return r;
  };
  __bf16* qt    = (__bf16*)take(32ull * 4096 * 512 * 2);  // q^T (n-major), bf16
  float*  kbuf  = (float*) take(32ull * 64 * 4096 * 4);   // raw k, f32
  __bf16* vbuf  = (__bf16*)take(32ull * 64 * 4096 * 2);   // v, bf16
  float*  kmax  = (float*) take(32ull * 64 * 4);
  float*  ksum  = (float*) take(32ull * 64 * 4);
  float*  ctxb  = (float*) take(32ull * 64 * 64 * 4);
  __bf16* wcb   = (__bf16*)take(32ull * 256 * 512 * 2);   // Wo ∘ ctx per batch
  __bf16* wqkv  = (__bf16*)take(640ull * 256 * 2);        // packed QKV weights
  float*  biasp = (float*) take(640ull * 4);              // packed bias

  pack_w_kernel<<<dim3(640), 256, 0, stream>>>(Wq, bq, Wk, bk, Wv, bv, wqkv, biasp);
  proj_qkv_kernel<<<dim3(5, 32, 32), 256, PROJ_SMEM, stream>>>(
      x, wqkv, biasp, qt, kbuf, vbuf);
  kstats_kernel<<<dim3(32 * 64), 256, 0, stream>>>(kbuf, kmax, ksum);
  ctx_kernel<<<dim3(32), 128, 0, stream>>>(kbuf, vbuf, kmax, ksum, ctxb);
  wc_kernel<<<dim3(8, 32), 256, 0, stream>>>(Wo, ctxb, wcb);
  out_proj_kernel<<<dim3(2, 32, 32), 256, OUTP_SMEM, stream>>>(wcb, qt, bo, out);
}
